// DumberTransducer_8272107012807
// MI455X (gfx1250) — matrix operand order repaired and checked by reading, not compiled, not run on hardware
//
#include <hip/hip_runtime.h>

// ---- problem constants ----
#define ALPHA 128
#define TAGD  64
#define LDIM  50
#define BDIM  4096
#define EMBD  40
#define CTXD  20
#define ATTD  100
#define FREQD 60
// 2*ln(10000)/60
#define KLOG  0.30701134573254044f

typedef _Float16 v16h __attribute__((ext_vector_type(16)));
typedef _Float16 v8h  __attribute__((ext_vector_type(8)));
typedef float    v8f  __attribute__((ext_vector_type(8)));

// TDM descriptor vector types (probe-verified builtin signatures)
typedef unsigned int tdm_v4u __attribute__((ext_vector_type(4)));
typedef int          tdm_v8i __attribute__((ext_vector_type(8)));
typedef int          tdm_v4i __attribute__((ext_vector_type(4)));

__device__ __forceinline__ float fastrcp(float x) { return __builtin_amdgcn_rcpf(x); }
__device__ __forceinline__ float sigmoidf_(float x) {
    return fastrcp(1.0f + __expf(-x));   // v_rcp_f32 instead of precise-div chain
}

// positional encoding value: element e of abs_pos[pos], FREQ=60
__device__ __forceinline__ float posenc(int pos, int e) {
    if (e < 30) return __sinf((float)pos * __expf(-(float)e * KLOG));
    return __cosf((float)pos * __expf(-(float)(e - 30) * KLOG));
}

// Load A fragment (16x32 f16) for this lane from a row-major LDS row.
// Per ISA: lane m = lane&15; halves: K in [khi, khi+8) and [16+khi, 16+khi+8)
// where khi = (lane>=16) ? 8 : 0.  Two 16-byte contiguous loads.
__device__ __forceinline__ v16h load_a(const _Float16* row, int k0) {
    int lane = threadIdx.x & 31;
    int khi = (lane & 16) ? 8 : 0;
    v8h lo = *(const v8h*)(row + k0 + khi);
    v8h hi = *(const v8h*)(row + k0 + 16 + khi);
    v16h a;
#pragma unroll
    for (int e = 0; e < 8; ++e) { a[e] = lo[e]; a[e + 8] = hi[e]; }
    return a;
}

// Load B fragment (32x16 f16): lane holds column n = lane&15; K in [khi, khi+8) and [16+khi, 16+khi+8).
__device__ __forceinline__ v16h load_b(const _Float16* Bmat, int ldb, int k0, int n0) {
    int lane = threadIdx.x & 31;
    int khi = (lane & 16) ? 8 : 0;
    const _Float16* col = Bmat + n0 + (lane & 15);
    v16h b;
#pragma unroll
    for (int e = 0; e < 8; ++e) {
        b[e] = col[(size_t)(k0 + khi + e) * ldb];
        b[e + 8] = col[(size_t)(k0 + 16 + khi + e) * ldb];
    }
    return b;
}

__device__ __forceinline__ v8f wmma_f16(v16h a, v16h b, v8f c) {
    return __builtin_amdgcn_wmma_f32_16x16x32_f16(false, a, false, b, (short)0, c,
                                                  false, false);
}

// TDM: async copy of a 2D f16 tile (rows x cols, row stride = cols) from global
// into LDS.  Descriptor per CDNA5 ISA D# spec (group0: count/lds/global/type,
// group1: data_size=2B, dims, tile, stride).  Tracked by TENSORcnt.
__device__ __forceinline__ void tdm_load_tile_f16(const _Float16* gsrc, void* lds_dst,
                                                  int rows, int cols) {
    unsigned long long ga = (unsigned long long)(uintptr_t)gsrc;
    unsigned int lds_off = (unsigned int)(uintptr_t)lds_dst;  // LDS aperture: low 32b
    tdm_v4u g0;
    g0[0] = 1u;                                   // count=1 (valid user descriptor)
    g0[1] = lds_off;                              // lds_addr [63:32]
    g0[2] = (unsigned int)ga;                     // global_addr [95:64]
    g0[3] = (unsigned int)((ga >> 32) & 0x1FFFFFFu) | (2u << 30);  // addr hi | type=2
    tdm_v8i g1;
    g1[0] = 0x00010000;                           // workgroup_mask=0, data_size=1 (2B)
    g1[1] = (cols << 16);                         // tensor_dim0[15:0] @ bits 63:48
    g1[2] = (rows << 16);                         // tensor_dim0 hi=0 | tensor_dim1[15:0]
    g1[3] = (cols << 16);                         // tensor_dim1 hi=0 | tile_dim0
    g1[4] = rows;                                 // tile_dim1 | tile_dim2=0
    g1[5] = cols;                                 // tensor_dim0_stride[31:0]
    g1[6] = 0;                                    // stride hi | tensor_dim1_stride lo
    g1[7] = 0;
    tdm_v4i z4 = {0, 0, 0, 0};
#if defined(__clang_major__) && (__clang_major__ >= 23)
    tdm_v8i z8 = {0, 0, 0, 0, 0, 0, 0, 0};
    __builtin_amdgcn_tensor_load_to_lds(g0, g1, z4, z4, z8, 0);
#else
    __builtin_amdgcn_tensor_load_to_lds(g0, g1, z4, z4, 0);
#endif
}

// ---------------- kernel 1: weight conversion to padded f16 ----------------
__global__ void prep_kernel(const float* __restrict__ W_q, const float* __restrict__ W_k,
                            const float* __restrict__ W_ff, const float* __restrict__ W_re,
                            const float* __restrict__ W_bc,
                            _Float16* Wq, _Float16* Wk, _Float16* Wff,
                            _Float16* Wre, _Float16* Wbc) {
    int tid = blockIdx.x * blockDim.x + threadIdx.x;
    int stride = gridDim.x * blockDim.x;
    // W_q [272][100] -> [288][112]
    for (int i = tid; i < 288 * 112; i += stride) {
        int k = i / 112, n = i % 112;
        Wq[i] = (k < 272 && n < 100) ? (_Float16)W_q[k * 100 + n] : (_Float16)0.f;
    }
    // W_k [120][100] -> [128][112]
    for (int i = tid; i < 128 * 112; i += stride) {
        int k = i / 112, n = i % 112;
        Wk[i] = (k < 120 && n < 100) ? (_Float16)W_k[k * 100 + n] : (_Float16)0.f;
    }
    // W_ff [420][128] -> [448][128]
    for (int i = tid; i < 448 * 128; i += stride) {
        int k = i >> 7, n = i & 127;
        Wff[i] = (k < 420) ? (_Float16)W_ff[k * 128 + n] : (_Float16)0.f;
    }
    // W_re [128][20] -> [128][32]
    for (int i = tid; i < 128 * 32; i += stride) {
        int k = i >> 5, n = i & 31;
        Wre[i] = (n < 20) ? (_Float16)W_re[k * 20 + n] : (_Float16)0.f;
    }
    // W_bc [20][20] -> [32][32]
    for (int i = tid; i < 32 * 32; i += stride) {
        int k = i >> 5, n = i & 31;
        Wbc[i] = (k < 20 && n < 20) ? (_Float16)W_bc[k * 20 + n] : (_Float16)0.f;
    }
}

// ---------------- kernel 2: ce = emb_table[lemma] @ W_ce + b_ce -------------
__global__ void enc_ce_kernel(const int* __restrict__ lemma,
                              const float* __restrict__ emb_table,
                              const float* __restrict__ W_ce, const float* __restrict__ b_ce,
                              float* __restrict__ ce) {
    size_t idx = (size_t)blockIdx.x * blockDim.x + threadIdx.x;
    if (idx >= (size_t)LDIM * BDIM) return;
    int letter = lemma[idx];
    const float* er = emb_table + (size_t)letter * EMBD;
    float e[EMBD];
#pragma unroll
    for (int k = 0; k < EMBD; ++k) e[k] = er[k];
    float* cr = ce + idx * CTXD;
#pragma unroll
    for (int j = 0; j < CTXD; ++j) {
        float s = b_ce[j];
#pragma unroll
        for (int k = 0; k < EMBD; ++k) s += e[k] * W_ce[k * CTXD + j];
        cr[j] = s;
    }
}

// ------- kernel 3: ctxt / V (VALU) then K = sigmoid(pos_stack@W_k) (WMMA) ---
// W_k tile is DMA'd into LDS by the Tensor Data Mover while phase 1 runs.
__global__ void __launch_bounds__(32)
enc_k_kernel(const int* __restrict__ lemma, const float* __restrict__ emb_table,
             const float* __restrict__ ce,
             const float* __restrict__ W_b4, const float* __restrict__ b_b4,
             const float* __restrict__ W_aft, const float* __restrict__ b_aft,
             const float* __restrict__ W_all, const float* __restrict__ b_all,
             const _Float16* __restrict__ Wk, const float* __restrict__ b_k,
             _Float16* __restrict__ Vout, _Float16* __restrict__ Kout) {
    __shared__ __align__(16) _Float16 kA[32][128];   // [pos(60)|ctxt(20)|emb(40)|pad(8)]
    __shared__ __align__(16) _Float16 wk[128 * 112]; // TDM-resident W_k tile
    const int lane = threadIdx.x;
    const size_t row0 = (size_t)blockIdx.x * 32;
    const size_t row = row0 + lane;          // flat (l*B + b)
    const int l = (int)(row / BDIM);
    const int b = (int)(row % BDIM);

    // kick off async weight tile load (TENSORcnt) before the VALU phase
    tdm_load_tile_f16(Wk, (void*)wk, 128, 112);

    // ---- phase 1: per-row VALU encoder ----
    {
        float in40[40];
#pragma unroll
        for (int k = 0; k < 40; ++k) in40[k] = 0.f;
        if (l >= 2) {
            const float* c2 = ce + ((size_t)(l - 2) * BDIM + b) * CTXD;
#pragma unroll
            for (int k = 0; k < CTXD; ++k) in40[k] = c2[k];
        }
        if (l >= 1) {
            const float* c1 = ce + ((size_t)(l - 1) * BDIM + b) * CTXD;
#pragma unroll
            for (int k = 0; k < CTXD; ++k) in40[20 + k] = c1[k];
        }
        float b4aft[40];
#pragma unroll
        for (int j = 0; j < CTXD; ++j) {
            float s = b_b4[j];
#pragma unroll
            for (int k = 0; k < 40; ++k) s += in40[k] * W_b4[k * CTXD + j];
            b4aft[j] = sigmoidf_(s);
        }
        float cp1[CTXD];
#pragma unroll
        for (int k = 0; k < CTXD; ++k) cp1[k] = 0.f;
        if (l + 1 < LDIM) {
            const float* c1 = ce + ((size_t)(l + 1) * BDIM + b) * CTXD;
#pragma unroll
            for (int k = 0; k < CTXD; ++k) cp1[k] = c1[k];
        }
#pragma unroll
        for (int j = 0; j < CTXD; ++j) {
            float s = b_aft[j];
#pragma unroll
            for (int k = 0; k < CTXD; ++k) s += cp1[k] * W_aft[k * CTXD + j];
            b4aft[20 + j] = sigmoidf_(s);
        }
#pragma unroll
        for (int j = 0; j < CTXD; ++j) {
            float s = b_all[j];
#pragma unroll
            for (int k = 0; k < 40; ++k) s += b4aft[k] * W_all[k * CTXD + j];
            float ct = sigmoidf_(s);
            kA[lane][60 + j] = (_Float16)ct;
            Vout[row * 60 + j] = (_Float16)ct;
        }
#pragma unroll
        for (int e = 0; e < 60; ++e) kA[lane][e] = (_Float16)posenc(l, e);
        int letter = lemma[row];
        const float* er = emb_table + (size_t)letter * EMBD;
#pragma unroll
        for (int k = 0; k < EMBD; ++k) {
            float ev = er[k];
            kA[lane][80 + k] = (_Float16)ev;
            Vout[row * 60 + 20 + k] = (_Float16)ev;
        }
#pragma unroll
        for (int c = 120; c < 128; ++c) kA[lane][c] = (_Float16)0.f;
    }
    __syncthreads();
    __builtin_amdgcn_s_wait_tensorcnt(0);   // W_k tile resident in LDS

    // ---- phase 2: K = sigmoid(A[16x128] @ wk[128x112] + b_k), 2 tiles ----
    const int nn = lane & 15;
    const int mb = (lane & 16) ? 8 : 0;
    for (int t = 0; t < 2; ++t) {
        const _Float16* Arow = &kA[t * 16 + (lane & 15)][0];
        for (int nt = 0; nt < 7; ++nt) {
            int n0 = nt * 16;
            v8f c = {};
            for (int k0 = 0; k0 < 128; k0 += 32)
                c = wmma_f16(load_a(Arow, k0), load_b(wk, 112, k0, n0), c);
            int n = n0 + nn;
            if (n < ATTD) {
                float bias = b_k[n];
#pragma unroll
                for (int r = 0; r < 8; ++r)
                    Kout[(row0 + t * 16 + mb + r) * ATTD + n] =
                        (_Float16)sigmoidf_(c[r] + bias);
            }
        }
    }
}

// -------- kernel 4: decoder scan, 16 batch rows per block, 4 waves ----------
__global__ void __launch_bounds__(128)
decoder_kernel(const float* __restrict__ tags,
               const _Float16* __restrict__ K, const _Float16* __restrict__ V,
               const _Float16* __restrict__ Wq, const float* __restrict__ b_q,
               const _Float16* __restrict__ Wff, const float* __restrict__ b_ff,
               const _Float16* __restrict__ Wre, const float* __restrict__ b_re,
               const _Float16* __restrict__ Wbc, const float* __restrict__ b_bc,
               float* __restrict__ out) {
    // qA row: [pos(0..59)|ctx(60..79)|probs(80..207)|tags(208..271)|pad(272..287)]
    // xA row: [in_attn(0..59)|zeros(60..207)|ctx(208..227)|probs(228..355)|tags(356..419)|pad..447]
    __shared__ __align__(16) _Float16 qA[16][288];
    __shared__ __align__(16) _Float16 xA[16][448];
    __shared__ __align__(16) _Float16 rA[16][32];   // reembed carry (f16, padded)
    __shared__ float qbuf[16][ATTD];
    __shared__ float att[16][52];
    __shared__ float pbuf[16][ALPHA];
    __shared__ float posv[64];
    __shared__ float row_inv[16];

    const int tid = threadIdx.x;
    const int wave = tid >> 5;
    const int lane = tid & 31;
    const int b0 = blockIdx.x * 16;

    // ---- init ----
    for (int idx = tid; idx < 16 * 288; idx += 128) ((_Float16*)qA)[idx] = (_Float16)0.f;
    for (int idx = tid; idx < 16 * 448; idx += 128) ((_Float16*)xA)[idx] = (_Float16)0.f;
    for (int idx = tid; idx < 16 * 32; idx += 128) ((_Float16*)rA)[idx] = (_Float16)0.f;
    for (int idx = tid; idx < 16 * TAGD; idx += 128) {
        int r = idx >> 6, c = idx & 63;
        _Float16 t = (_Float16)tags[(size_t)(b0 + r) * TAGD + c];
        qA[r][208 + c] = t;
        xA[r][356 + c] = t;
    }
    // probs[0] = one-hot START=1; write out[0]
    for (int idx = tid; idx < 16 * ALPHA; idx += 128) {
        int r = idx >> 7, a = idx & 127;
        float p = (a == 1) ? 1.f : 0.f;
        out[(size_t)(b0 + r) * ALPHA + a] = p;
        qA[r][80 + a] = (_Float16)p;
        xA[r][228 + a] = (_Float16)p;
    }
    __syncthreads();

    const _Float16* Arow_q = &qA[lane & 15][0];
    const _Float16* Arow_x = &xA[lane & 15][0];
    const _Float16* Arow_r = &rA[lane & 15][0];
    const int nn = lane & 15;
    const int mb = (lane & 16) ? 8 : 0;
    const int r16 = tid & 15;
    const int slice = tid >> 4;   // 0..7

    for (int i = 1; i < LDIM; ++i) {
        // pos_i broadcast into qA[:, 0..59]
        if (tid < 60) posv[tid] = posenc(i, tid);
        __syncthreads();
        for (int idx = tid; idx < 16 * 64; idx += 128) {
            int r = idx >> 6, c = idx & 63;
            if (c < 60) qA[r][c] = (_Float16)posv[c];
        }
        __syncthreads();

        // ctx = sigmoid(r_pp @ W_bc + b_bc): waves 0..1 take one n-tile each
        for (int nt = wave; nt < 2; nt += 4) {
            int n0 = nt * 16;
            v8f c = {};
            c = wmma_f16(load_a(Arow_r, 0), load_b(Wbc, 32, 0, n0), c);
            int n = n0 + nn;
            if (n < CTXD) {
                float bias = b_bc[n];
#pragma unroll
                for (int r = 0; r < 8; ++r) {
                    _Float16 h = (_Float16)sigmoidf_(c[r] + bias);
                    qA[mb + r][60 + n] = h;
                    xA[mb + r][208 + n] = h;
                }
            }
        }
        __syncthreads();

        // q = sigmoid(qA[16x288] @ Wq[288x112] + b_q), n-tiles split over waves
        for (int nt = wave; nt < 7; nt += 4) {
            int n0 = nt * 16;
            v8f c = {};
            for (int k0 = 0; k0 < 288; k0 += 32)
                c = wmma_f16(load_a(Arow_q, k0), load_b(Wq, 112, k0, n0), c);
            int n = n0 + nn;
            if (n < ATTD) {
                float bias = b_q[n];
#pragma unroll
                for (int r = 0; r < 8; ++r) qbuf[mb + r][n] = sigmoidf_(c[r] + bias);
            }
        }
        __syncthreads();

        // scores[r][l] = q[r] . K[l, b0+r]   (128 lanes: 16 rows x 8 l-slices)
        {
            int lend = slice * 7 + 7;
            if (lend > LDIM) lend = LDIM;
            for (int l = slice * 7; l < lend; ++l) {
                const _Float16* kr = K + ((size_t)l * BDIM + b0 + r16) * ATTD;
                float s = 0.f;
#pragma unroll 4
                for (int a2 = 0; a2 < ATTD; ++a2) s += qbuf[r16][a2] * (float)kr[a2];
                att[r16][l] = s;
            }
        }
        __syncthreads();
        // softmax over l
        if (tid < 16) {
            float mx = att[tid][0];
            for (int l = 1; l < LDIM; ++l) mx = fmaxf(mx, att[tid][l]);
            float sum = 0.f;
            for (int l = 0; l < LDIM; ++l) {
                float e = __expf(att[tid][l] - mx);
                att[tid][l] = e;
                sum += e;
            }
            float inv = fastrcp(sum);
            for (int l = 0; l < LDIM; ++l) att[tid][l] *= inv;
        }
        __syncthreads();
        // in_attn[r][d] = sum_l attn[r][l] * V[l, b0+r, d]  (16 rows x 8 d-slices)
        {
            int d0 = slice * 8;
            int dn = 60 - d0;
            if (dn > 8) dn = 8;
            float acc[8];
#pragma unroll
            for (int d = 0; d < 8; ++d) acc[d] = 0.f;
            for (int l = 0; l < LDIM; ++l) {
                float w = att[r16][l];
                const _Float16* vr = V + ((size_t)l * BDIM + b0 + r16) * 60 + d0;
#pragma unroll
                for (int d = 0; d < 8; ++d)
                    if (d < dn) acc[d] += w * (float)vr[d];
            }
            for (int d = 0; d < dn; ++d) xA[r16][d0 + d] = (_Float16)acc[d];
        }
        __syncthreads();

        // ff = sigmoid(xA[16x448] @ Wff[448x128] + b_ff), split over waves
        for (int nt = wave; nt < 8; nt += 4) {
            int n0 = nt * 16;
            v8f c = {};
            for (int k0 = 0; k0 < 448; k0 += 32)
                c = wmma_f16(load_a(Arow_x, k0), load_b(Wff, 128, k0, n0), c);
            int n = n0 + nn;
            float bias = b_ff[n];
#pragma unroll
            for (int r = 0; r < 8; ++r) pbuf[mb + r][n] = sigmoidf_(c[r] + bias);
        }
        // r_new = sigmoid(probs_prev @ W_re + b_re)  (probs still intact in qA)
        for (int nt = wave; nt < 2; nt += 4) {
            int n0 = nt * 16;
            v8f c = {};
            for (int k0 = 0; k0 < 128; k0 += 32)
                c = wmma_f16(load_a(Arow_q + 80, k0), load_b(Wre, 32, k0, n0), c);
            int n = n0 + nn;
            if (n < CTXD) {
                float bias = b_re[n];
#pragma unroll
                for (int r = 0; r < 8; ++r)
                    rA[mb + r][n] = (_Float16)sigmoidf_(c[r] + bias);
            }
        }
        __syncthreads();

        // p = softmax(ff): rows by lanes 0..15, then block-wide normalize+store
        if (tid < 16) {
            float mx = pbuf[tid][0];
            for (int a2 = 1; a2 < ALPHA; ++a2) mx = fmaxf(mx, pbuf[tid][a2]);
            float sum = 0.f;
            for (int a2 = 0; a2 < ALPHA; ++a2) {
                float e = __expf(pbuf[tid][a2] - mx);
                pbuf[tid][a2] = e;
                sum += e;
            }
            row_inv[tid] = fastrcp(sum);
        }
        __syncthreads();
        for (int idx = tid; idx < 16 * ALPHA; idx += 128) {
            int r = idx >> 7, a = idx & 127;
            float p = pbuf[r][a] * row_inv[r];
            out[(size_t)i * BDIM * ALPHA + (size_t)(b0 + r) * ALPHA + a] = p;
            _Float16 h = (_Float16)p;
            qA[r][80 + a] = h;
            xA[r][228 + a] = h;
        }
        __syncthreads();
    }
}

extern "C" void kernel_launch(void* const* d_in, const int* in_sizes, int n_in,
                              void* d_out, int out_size, void* d_ws, size_t ws_size,
                              hipStream_t stream) {
    (void)in_sizes; (void)n_in; (void)out_size; (void)ws_size;
    const int*   lemma     = (const int*)d_in[0];
    const float* tags      = (const float*)d_in[1];
    const float* emb_table = (const float*)d_in[2];
    const float* W_ce  = (const float*)d_in[3];  const float* b_ce  = (const float*)d_in[4];
    const float* W_b4  = (const float*)d_in[5];  const float* b_b4  = (const float*)d_in[6];
    const float* W_aft = (const float*)d_in[7];  const float* b_aft = (const float*)d_in[8];
    const float* W_all = (const float*)d_in[9];  const float* b_all = (const float*)d_in[10];
    const float* W_re  = (const float*)d_in[11]; const float* b_re  = (const float*)d_in[12];
    const float* W_bc  = (const float*)d_in[13]; const float* b_bc  = (const float*)d_in[14];
    const float* W_q   = (const float*)d_in[15]; const float* b_q   = (const float*)d_in[16];
    const float* W_k   = (const float*)d_in[17]; const float* b_k   = (const float*)d_in[18];
    const float* W_ff  = (const float*)d_in[19]; const float* b_ff  = (const float*)d_in[20];
    float* out = (float*)d_out;

    // workspace carve-up (256B aligned regions)
    char* p = (char*)d_ws;
    auto carve = [&](size_t bytes) -> void* {
        void* r = (void*)p;
        p += (bytes + 255) & ~(size_t)255;
        return r;
    };
    float*    ce    = (float*)carve((size_t)LDIM * BDIM * CTXD * sizeof(float));
    _Float16* Vbuf  = (_Float16*)carve((size_t)LDIM * BDIM * 60 * sizeof(_Float16));
    _Float16* Kbuf  = (_Float16*)carve((size_t)LDIM * BDIM * ATTD * sizeof(_Float16));
    _Float16* Wq16  = (_Float16*)carve((size_t)288 * 112 * sizeof(_Float16));
    _Float16* Wk16  = (_Float16*)carve((size_t)128 * 112 * sizeof(_Float16));
    _Float16* Wff16 = (_Float16*)carve((size_t)448 * 128 * sizeof(_Float16));
    _Float16* Wre16 = (_Float16*)carve((size_t)128 * 32 * sizeof(_Float16));
    _Float16* Wbc16 = (_Float16*)carve((size_t)32 * 32 * sizeof(_Float16));

    prep_kernel<<<128, 256, 0, stream>>>(W_q, W_k, W_ff, W_re, W_bc,
                                         Wq16, Wk16, Wff16, Wre16, Wbc16);
    enc_ce_kernel<<<(LDIM * BDIM) / 256, 256, 0, stream>>>(lemma, emb_table, W_ce, b_ce, ce);
    enc_k_kernel<<<(LDIM * BDIM) / 32, 32, 0, stream>>>(lemma, emb_table, ce,
                                                        W_b4, b_b4, W_aft, b_aft,
                                                        W_all, b_all, Wk16, b_k,
                                                        Vbuf, Kbuf);
    decoder_kernel<<<BDIM / 16, 128, 0, stream>>>(tags, Kbuf, Vbuf,
                                                  Wq16, b_q, Wff16, b_ff,
                                                  Wre16, b_re, Wbc16, b_bc, out);
}
